// WindowCrossAttention3D_42880953483908
// MI455X (gfx1250) — compile-verified, matrix operand order checked by reading, not run on hardware
//
#include <hip/hip_runtime.h>
#include <hip/hip_bf16.h>

typedef _Float16 v16h __attribute__((ext_vector_type(16)));
typedef _Float16 v8h  __attribute__((ext_vector_type(8)));
typedef float    v8f  __attribute__((ext_vector_type(8)));
typedef float    v4f  __attribute__((ext_vector_type(4)));

#define MAX_LOG_SCALE 4.605170185988092f  // log(100)

// ---------------------------------------------------------------------------
// Weight packing: f32 row-major W[R][128] -> f16 WMMA B-fragments.
// Fragment (nt,kt) covers output cols [nt*16,nt*16+16), K cols [kt*32,kt*32+32).
// Per-fragment layout: 32 lanes x 16 halves, lane-major, so the main kernel
// loads a lane's v16h with two coalesced global_load_b128.
// B element for (lane,half): n = lane&15, 16-bit WMMA K packing:
//   kgroup = (lane<16)?0:8 ; v=half/2, hh=half%2
//   k = (v<4) ? kgroup+2v+hh : 16+kgroup+2(v-4)+hh
// ws layout (halves): qw[32 frags] | kvw[64 frags] | projw[32 frags], 512 each.
// One thread per (frag,lane): computes 16 halves, 2x global_store_b128.
// ---------------------------------------------------------------------------
__global__ void wca3d_pack_weights(const float* __restrict__ qw,
                                   const float* __restrict__ kvw,
                                   const float* __restrict__ pw,
                                   _Float16* __restrict__ ws) {
  int idx  = blockIdx.x * 256 + threadIdx.x;      // 0 .. 4095
  int lane = idx & 31;
  int frag = idx >> 5;                            // 0 .. 127
  const float* W;
  int base;
  if (frag < 32)      { W = qw;  base = frag; }
  else if (frag < 96) { W = kvw; base = frag - 32; }
  else                { W = pw;  base = frag - 96; }
  int kt = base & 3;
  int nt = base >> 2;
  int n    = lane & 15;
  int kgrp = (lane < 16) ? 0 : 8;
  int row  = nt * 16 + n;
  const float* Wr = W + row * 128 + kt * 32;
  v8h o0, o1;
#pragma unroll
  for (int half = 0; half < 8; ++half) {
    int v = half >> 1, hh = half & 1;
    int k = kgrp + 2 * v + hh;                    // VGPR 0..3 block
    o0[half] = (_Float16)Wr[k];
  }
#pragma unroll
  for (int half = 0; half < 8; ++half) {
    int v = half >> 1, hh = half & 1;
    int k = 16 + kgrp + 2 * v + hh;               // VGPR 4..7 block
    o1[half] = (_Float16)Wr[k];
  }
  _Float16* dst = ws + (size_t)idx * 16;
  *(v8h*)dst       = o0;
  *(v8h*)(dst + 8) = o1;
}

// ---------------------------------------------------------------------------
// Fragment loaders
// ---------------------------------------------------------------------------
__device__ inline v16h load_frag2(const _Float16* p0, const _Float16* p1) {
  v8h lo = *(const v8h*)p0;
  v8h hi = *(const v8h*)p1;
  v16h r;
#pragma unroll
  for (int i = 0; i < 8; ++i) { r[i] = lo[i]; r[8 + i] = hi[i]; }
  return r;
}

__device__ inline v8f wmma_f16(v16h a, v16h b, v8f c) {
  return __builtin_amdgcn_wmma_f32_16x16x32_f16(false, a, false, b,
                                                (short)0, c, false, false);
}

// ---------------------------------------------------------------------------
// Main kernel: one block per window. 256 threads = 8 wave32s.
// LDS (64KB, aliased):
//   buf0: x1 f16 (64x128)          -> later V^T (128 cols x 64 tokens)
//   buf1: x2 f16 (64x128)          -> later O   (64x128)
//   buf2: Q  f16 (64x128, normed)  -> later P heads 0,1 (2 x 64x64)
//   buf3: K  f16 (64x128, normed)  -> later P heads 2,3
// Streaming data (x1/x2 in, out) uses non-temporal hints so the 768MB
// activation stream does not evict the L2-resident packed weights.
// ---------------------------------------------------------------------------
__global__ __launch_bounds__(256)
void wca3d_main(const float* __restrict__ x1, const float* __restrict__ x2,
                const float* __restrict__ qb, const float* __restrict__ vb,
                const float* __restrict__ ls, const float* __restrict__ pbias,
                const _Float16* __restrict__ wpack, float* __restrict__ out) {
  __shared__ _Float16 sm[32768];
  _Float16* buf0 = sm;
  _Float16* buf1 = sm + 8192;
  _Float16* buf2 = sm + 16384;
  _Float16* buf3 = sm + 24576;

  const int tid    = threadIdx.x;
  const int lane   = tid & 31;
  const int wave   = tid >> 5;
  const int nlane  = lane & 15;
  const int kgrp   = (lane < 16) ? 0 : 8;
  const int rowoff = (lane < 16) ? 0 : 8;
  const int b      = blockIdx.x;

  const _Float16* qwp  = wpack;           // 32 frags * 512 halves
  const _Float16* kvwp = wpack + 16384;   // 64 frags
  const _Float16* pwp  = wpack + 49152;   // 32 frags

  // ---- Phase 0: load x1,x2 (NT) -> LDS f16 (b128 LDS stores) --------------
  {
    const v4f* x1v = (const v4f*)x1 + (size_t)b * 2048;
    const v4f* x2v = (const v4f*)x2 + (size_t)b * 2048;
#pragma unroll
    for (int j = 0; j < 4; ++j) {
      int i = tid + j * 256;              // 8 halves per thread per iter
      v4f a0 = __builtin_nontemporal_load(&x1v[i * 2]);
      v4f a1 = __builtin_nontemporal_load(&x1v[i * 2 + 1]);
      v8h h;
#pragma unroll
      for (int e = 0; e < 4; ++e) { h[e] = (_Float16)a0[e]; h[4 + e] = (_Float16)a1[e]; }
      *(v8h*)&buf0[i * 8] = h;
      v4f c0 = __builtin_nontemporal_load(&x2v[i * 2]);
      v4f c1 = __builtin_nontemporal_load(&x2v[i * 2 + 1]);
      v8h h2;
#pragma unroll
      for (int e = 0; e < 4; ++e) { h2[e] = (_Float16)c0[e]; h2[4 + e] = (_Float16)c1[e]; }
      *(v8h*)&buf1[i * 8] = h2;
    }
  }
  __syncthreads();

  // ---- Phase 1a: Q = X1 @ qw^T + qb -> buf2 (row-major) -------------------
#pragma unroll
  for (int i = 0; i < 4; ++i) {
    int t = wave * 4 + i;
    int mt = t >> 3, nt = t & 7;
    v8f acc = {0, 0, 0, 0, 0, 0, 0, 0};
#pragma unroll
    for (int kt = 0; kt < 4; ++kt) {
      const _Float16* ar = &buf0[(mt * 16 + nlane) * 128 + kt * 32];
      v16h a = load_frag2(ar + kgrp, ar + 16 + kgrp);
      const _Float16* wp = &qwp[(nt * 4 + kt) * 512 + lane * 16];
      v16h bf = load_frag2(wp, wp + 8);
      acc = wmma_f16(a, bf, acc);
    }
    int c = nt * 16 + nlane;
    float bias = qb[c];
#pragma unroll
    for (int r = 0; r < 8; ++r) {
      int m = mt * 16 + rowoff + r;
      buf2[m * 128 + c] = (_Float16)(acc[r] + bias);
    }
  }
  __syncthreads();  // x1 (buf0) dead; V^T may now use buf0

  // ---- Phase 1b: KV = X2 @ kvw^T; K -> buf3 row-major, V(+vb) -> buf0 as V^T
#pragma unroll
  for (int i = 0; i < 8; ++i) {
    int t = wave * 8 + i;
    int mt = t >> 4, nt = t & 15;
    v8f acc = {0, 0, 0, 0, 0, 0, 0, 0};
#pragma unroll
    for (int kt = 0; kt < 4; ++kt) {
      const _Float16* ar = &buf1[(mt * 16 + nlane) * 128 + kt * 32];
      v16h a = load_frag2(ar + kgrp, ar + 16 + kgrp);
      const _Float16* wp = &kvwp[(nt * 4 + kt) * 512 + lane * 16];
      v16h bf = load_frag2(wp, wp + 8);
      acc = wmma_f16(a, bf, acc);
    }
    if (nt < 8) {                       // K part (bias = 0), row-major
      int c = nt * 16 + nlane;
#pragma unroll
      for (int r = 0; r < 8; ++r) {
        int m = mt * 16 + rowoff + r;
        buf3[m * 128 + c] = (_Float16)acc[r];
      }
    } else {                            // V part: transposed, rows contiguous
      int c = (nt - 8) * 16 + nlane;    //   per lane -> one ds_store_b128
      float bias = vb[c];
      v8h pk;
#pragma unroll
      for (int r = 0; r < 8; ++r) pk[r] = (_Float16)(acc[r] + bias);
      *(v8h*)&buf0[c * 64 + mt * 16 + rowoff] = pk;
    }
  }
  __syncthreads();

  // ---- Phase 2: L2-normalize q (fold in logit scale) and k (vectorized) ---
  {
    int token = tid & 63;
    int head  = tid >> 6;
    float sc = __expf(fminf(ls[head], MAX_LOG_SCALE));

    _Float16* qrow = &buf2[token * 128 + head * 32];
    v8h qv[4];
    float s = 0.f;
#pragma unroll
    for (int c = 0; c < 4; ++c) {
      qv[c] = *(v8h*)&qrow[c * 8];
#pragma unroll
      for (int j = 0; j < 8; ++j) { float x = (float)qv[c][j]; s += x * x; }
    }
    float rq = sc / fmaxf(sqrtf(s), 1e-12f);
#pragma unroll
    for (int c = 0; c < 4; ++c) {
#pragma unroll
      for (int j = 0; j < 8; ++j) qv[c][j] = (_Float16)((float)qv[c][j] * rq);
      *(v8h*)&qrow[c * 8] = qv[c];
    }

    _Float16* krow = &buf3[token * 128 + head * 32];
    v8h kv[4];
    s = 0.f;
#pragma unroll
    for (int c = 0; c < 4; ++c) {
      kv[c] = *(v8h*)&krow[c * 8];
#pragma unroll
      for (int j = 0; j < 8; ++j) { float x = (float)kv[c][j]; s += x * x; }
    }
    float rk = 1.0f / fmaxf(sqrtf(s), 1e-12f);
#pragma unroll
    for (int c = 0; c < 4; ++c) {
#pragma unroll
      for (int j = 0; j < 8; ++j) kv[c][j] = (_Float16)((float)kv[c][j] * rk);
      *(v8h*)&krow[c * 8] = kv[c];
    }
  }
  __syncthreads();

  // ---- Phase 3: S = qn @ kn^T per head; wave owns full rows ---------------
  int h     = wave >> 1;
  int mbase = (wave & 1) * 2;
  v8f sacc[2][4];
#pragma unroll
  for (int mi = 0; mi < 2; ++mi) {
    int mt = mbase + mi;
#pragma unroll
    for (int nt = 0; nt < 4; ++nt) {
      const _Float16* ar = &buf2[(mt * 16 + nlane) * 128 + h * 32];
      v16h a = load_frag2(ar + kgrp, ar + 16 + kgrp);
      const _Float16* br = &buf3[(nt * 16 + nlane) * 128 + h * 32];
      v16h bf = load_frag2(br + kgrp, br + 16 + kgrp);
      v8f z = {0, 0, 0, 0, 0, 0, 0, 0};
      sacc[mi][nt] = wmma_f16(a, bf, z);
    }
  }
  __syncthreads();  // done reading q/k; their LDS becomes P storage

  // ---- softmax (registers) + write P f16 ----------------------------------
  _Float16* pbase = (h < 2) ? (buf2 + h * 4096) : (buf3 + (h - 2) * 4096);
#pragma unroll
  for (int mi = 0; mi < 2; ++mi) {
    int mt = mbase + mi;
#pragma unroll
    for (int r = 0; r < 8; ++r) {
      float mx = fmaxf(fmaxf(sacc[mi][0][r], sacc[mi][1][r]),
                       fmaxf(sacc[mi][2][r], sacc[mi][3][r]));
#pragma unroll
      for (int off = 1; off < 16; off <<= 1) mx = fmaxf(mx, __shfl_xor(mx, off, 16));
      float p0 = __expf(sacc[mi][0][r] - mx);
      float p1 = __expf(sacc[mi][1][r] - mx);
      float p2 = __expf(sacc[mi][2][r] - mx);
      float p3 = __expf(sacc[mi][3][r] - mx);
      float sum = p0 + p1 + p2 + p3;
#pragma unroll
      for (int off = 1; off < 16; off <<= 1) sum += __shfl_xor(sum, off, 16);
      float inv = 1.0f / sum;
      int m = mt * 16 + rowoff + r;
      pbase[m * 64 +  0 + nlane] = (_Float16)(p0 * inv);
      pbase[m * 64 + 16 + nlane] = (_Float16)(p1 * inv);
      pbase[m * 64 + 32 + nlane] = (_Float16)(p2 * inv);
      pbase[m * 64 + 48 + nlane] = (_Float16)(p3 * inv);
    }
  }
  __syncthreads();

  // ---- Phase 4: O = P @ V -> buf1 (x2 dead) -------------------------------
  // Prefetch this wave's proj-weight fragments for phase 5 (L2 -> near cache).
#pragma unroll
  for (int i = 0; i < 4; ++i) {
    int nt5 = (wave * 4 + i) & 7;
    __builtin_prefetch(&pwp[(nt5 * 4) * 512 + lane * 16], 0, 0);
  }
#pragma unroll
  for (int i = 0; i < 4; ++i) {
    int t = wave * 4 + i;
    int ho = t >> 3, rr = t & 7, mt = rr >> 1, ntv = rr & 1;
    const _Float16* pb_ = (ho < 2) ? (buf2 + ho * 4096) : (buf3 + (ho - 2) * 4096);
    int c = ho * 32 + ntv * 16 + nlane;
    v8f acc = {0, 0, 0, 0, 0, 0, 0, 0};
#pragma unroll
    for (int kt = 0; kt < 2; ++kt) {
      const _Float16* ar = &pb_[(mt * 16 + nlane) * 64 + kt * 32];
      v16h a = load_frag2(ar + kgrp, ar + 16 + kgrp);
      const _Float16* br = &buf0[c * 64 + kt * 32];   // V^T: [col][token]
      v16h bf = load_frag2(br + kgrp, br + 16 + kgrp);
      acc = wmma_f16(a, bf, acc);
    }
#pragma unroll
    for (int r = 0; r < 8; ++r) {
      int m = mt * 16 + rowoff + r;
      buf1[m * 128 + c] = (_Float16)acc[r];
    }
  }
  __syncthreads();

  // ---- Phase 5: out = O @ proj_w^T + proj_b (NT f32 global store) ---------
#pragma unroll
  for (int i = 0; i < 4; ++i) {
    int t = wave * 4 + i;
    int mt = t >> 3, nt = t & 7;
    v8f acc = {0, 0, 0, 0, 0, 0, 0, 0};
#pragma unroll
    for (int kt = 0; kt < 4; ++kt) {
      const _Float16* ar = &buf1[(mt * 16 + nlane) * 128 + kt * 32];
      v16h a = load_frag2(ar + kgrp, ar + 16 + kgrp);
      const _Float16* wp = &pwp[(nt * 4 + kt) * 512 + lane * 16];
      v16h bf = load_frag2(wp, wp + 8);
      acc = wmma_f16(a, bf, acc);
    }
    int c = nt * 16 + nlane;
    float bias = pbias[c];
    float* orow = out + (size_t)b * 8192;
#pragma unroll
    for (int r = 0; r < 8; ++r) {
      int m = mt * 16 + rowoff + r;
      __builtin_nontemporal_store(acc[r] + bias, &orow[m * 128 + c]);
    }
  }
}

// ---------------------------------------------------------------------------
extern "C" void kernel_launch(void* const* d_in, const int* in_sizes, int n_in,
                              void* d_out, int out_size, void* d_ws, size_t ws_size,
                              hipStream_t stream) {
  const float* x1  = (const float*)d_in[0];
  const float* x2  = (const float*)d_in[1];
  const float* qw  = (const float*)d_in[2];
  const float* qb  = (const float*)d_in[3];
  const float* kvw = (const float*)d_in[4];
  const float* vb  = (const float*)d_in[5];
  const float* ls  = (const float*)d_in[6];
  const float* pw  = (const float*)d_in[7];
  const float* pb  = (const float*)d_in[8];
  float* out = (float*)d_out;
  _Float16* wpack = (_Float16*)d_ws;   // needs 131072 bytes

  // Pack all three weight matrices into WMMA B-fragment layout (f16).
  wca3d_pack_weights<<<16, 256, 0, stream>>>(qw, kvw, pw, wpack);

  int nwin = in_sizes[0] / (64 * 128);   // 8192 windows
  wca3d_main<<<nwin, 256, 0, stream>>>(x1, x2, qb, vb, ls, pb, wpack, out);
}
